// SelfAttention_11802570129972
// MI455X (gfx1250) — matrix operand order repaired and checked by reading, not compiled, not run on hardware
//
#include <hip/hip_runtime.h>
#include <hip/hip_bf16.h>

// ---------------------------------------------------------------------------
// Self-attention, N=8192, D_IN=D_H=1024, fp32 in/out.  MI455X (gfx1250).
//   * ~327 GFLOP vs ~125 MB HBM traffic -> matrix-unit bound:
//     v_wmma_f32_16x16x32_bf16 (fp32 accumulate) for all three GEMM stages.
//   * Flash attention (never materialize the 256 MB score matrix); scratch =
//     q,k row-major bf16 + v^T bf16 = 48 MB; k+v (32 MB) are L2-resident
//     (192 MB L2) and streamed as b128 fragments.
//   * q tile staged in LDS once per block (async-to-LDS when available),
//     killing ~4 GB of redundant q re-reads from L2; QK A-frags become
//     ds_load_b128 on a bank-conflict-free padded layout.
// ---------------------------------------------------------------------------

#define NTOK 8192
#define DIN  1024
#define DH   1024

typedef __bf16 bf16;
typedef __attribute__((ext_vector_type(16))) __bf16 v16bf;
typedef __attribute__((ext_vector_type(8)))  float  v8f;
typedef __attribute__((ext_vector_type(4)))  int    v4i;

union FragBF {
    v16bf v;
    bf16  e[16];
    uint4 u[2];
};
static_assert(sizeof(FragBF) == 32, "bf16x16 fragment must be 8 VGPRs");

__device__ __forceinline__ v8f wmma_bf16(v16bf a, v16bf b, v8f c) {
    // D = A(16x32 bf16) x B(32x16 bf16) + C(16x16 f32)
    return __builtin_amdgcn_wmma_f32_16x16x32_bf16(
        false, a, false, b, (short)0, c, false, false);
}

__device__ __forceinline__ v8f zero8() {
    v8f z = {0.f, 0.f, 0.f, 0.f, 0.f, 0.f, 0.f, 0.f};
    return z;
}

// -------- async global->LDS copy (CDNA5 GLOBAL_LOAD_ASYNC_TO_LDS_B128) ------
#if defined(__has_builtin)
#if __has_builtin(__builtin_amdgcn_global_load_async_to_lds_b128)
#define HAVE_ASYNC_LDS 1
#endif
#endif

#ifdef HAVE_ASYNC_LDS
__device__ __forceinline__ void async_copy_b128(const bf16* g, bf16* l) {
    // ROCm 7.2 declares the builtin with generic v4i* params.
    // LDS pointer value: low 32 bits of the generic address are the LDS
    // offset (ISA 10.2).
    __builtin_amdgcn_global_load_async_to_lds_b128(
        (v4i*)(uintptr_t)g,
        (v4i*)(uintptr_t)(unsigned)(uintptr_t)l, 0, 0);
}
#endif

// ---------------------------------------------------------------------------
// Kernel 1: fused projections.  out = h @ W^T for W in {wq, wk, wv}.
// One wave per 16x16 tile; fp32 loaded b128-wide, cvt_pk to bf16 in regs.
// sel==2 (v) stores transposed (contiguous 8-elem store per lane from the
// C-matrix layout), so the P@V B-operand later reads contiguous v^T rows.
// ---------------------------------------------------------------------------
__global__ __launch_bounds__(256, 2)
void proj_kernel(const float* __restrict__ h,  const float* __restrict__ wq,
                 const float* __restrict__ wk, const float* __restrict__ wv,
                 bf16* __restrict__ qb, bf16* __restrict__ kb,
                 bf16* __restrict__ vt)
{
    const int wave = threadIdx.x >> 5;
    const int lane = threadIdx.x & 31;
    const int half = lane >> 4;
    const int ln   = lane & 15;

    const int mt  = blockIdx.x;              // row tile of h
    const int nt  = blockIdx.y * 8 + wave;   // col tile of output
    const int sel = blockIdx.z;              // 0=q 1=k 2=v

    const float* w  = (sel == 0) ? wq : (sel == 1) ? wk : wv;
    const float* ap = h + (long)(mt * 16 + ln) * DIN;
    const float* bp = w + (long)(nt * 16 + ln) * DIN;

    v8f acc = zero8();

#pragma unroll 4
    for (int kk = 0; kk < DIN; kk += 32) {
        FragBF a, b;
        const float* af  = ap + kk + half * 8;
        const float* bpf = bp + kk + half * 8;
#pragma unroll
        for (int i = 0; i < 8; ++i) {
            a.e[i]     = (bf16)af[i];
            a.e[i + 8] = (bf16)af[i + 16];
        }
#pragma unroll
        for (int i = 0; i < 8; ++i) {
            b.e[i]     = (bf16)bpf[i];
            b.e[i + 8] = (bf16)bpf[i + 16];
        }
        acc = wmma_bf16(a.v, b.v, acc);
    }

    if (sel < 2) {
        bf16* dst = (sel == 0) ? qb : kb;    // row-major [N][DH]
#pragma unroll
        for (int g = 0; g < 8; ++g)
            dst[(long)(mt * 16 + half * 8 + g) * DH + nt * 16 + ln] =
                (bf16)acc[g];
    } else {
        bf16* dst = vt + (long)(nt * 16 + ln) * NTOK + mt * 16 + half * 8;
#pragma unroll
        for (int g = 0; g < 8; ++g) dst[g] = (bf16)acc[g];
    }
}

// ---------------------------------------------------------------------------
// Kernel 2: flash attention.  Block = 32 q-rows x full 1024 output cols.
// 8 waves: mw = wave&1 -> 16-row subtile, jw = wave>>1 -> 256-col slab
// (16 v8f accumulators = 128 VGPRs).  Per 128-col KV chunk:
//   QK  : 2 score tiles/wave, A-frags from LDS q tile, B-frags stream k (L2)
//   stats: float4 LDS scans, online max/sum (v_exp co-executes with XDL)
//   PV  : 4 bf16 P-frags x 16 v^T B-frags -> 64 wmma
// ---------------------------------------------------------------------------
#define QPAD 1032   // row stride: 516 words == 4 banks -> conflict-free frags
#define SPAD 132

__global__ __launch_bounds__(256, 1)
void attn_kernel(const bf16* __restrict__ qb, const bf16* __restrict__ kb,
                 const bf16* __restrict__ vt, float* __restrict__ out)
{
    __shared__ bf16  q_lds[32][QPAD];        // 66,048 B
    __shared__ float S[32][SPAD];            // 16,896 B

    const int tid  = threadIdx.x;
    const int wave = tid >> 5;
    const int lane = tid & 31;
    const int half = lane >> 4;
    const int ln   = lane & 15;
    const int mw   = wave & 1;
    const int jw   = wave >> 1;
    const long block_m = (long)blockIdx.x * 32;
    const float scale = 0.03125f;            // 1/sqrt(1024)

    // ---- stage the 32x1024 bf16 q tile into LDS (async path if available)
    for (int it = tid; it < 32 * 128; it += 256) {
        const int row = it >> 7;             // 128 x 16B segments per row
        const int cc  = it & 127;
        const bf16* g = qb + (block_m + row) * DH + cc * 8;
        bf16*       l = &q_lds[row][cc * 8];
#ifdef HAVE_ASYNC_LDS
        async_copy_b128(g, l);
#else
        *(uint4*)l = *(const uint4*)g;
#endif
    }
#ifdef HAVE_ASYNC_LDS
    asm volatile("s_wait_asynccnt 0" ::: "memory");
#endif

    v8f acc[16];
#pragma unroll
    for (int t = 0; t < 16; ++t) acc[t] = zero8();

    float m_run = -3.0e38f;
    float l_run = 0.f;

    const bf16* qrow = &q_lds[mw * 16 + ln][0];

    for (int nc = 0; nc < NTOK; nc += 128) {
        __syncthreads();                     // q tile ready / prev S consumed

        // ---- S = (q @ k^T) * scale : two 16x16 score tiles per wave
#pragma unroll
        for (int st = 0; st < 2; ++st) {
            const bf16* krow =
                kb + (long)(nc + (jw * 2 + st) * 16 + ln) * DH;
            if (nc + 128 < NTOK)             // pull next chunk's row into L2
                __builtin_prefetch(krow + (size_t)128 * DH, 0, 0);
            v8f s = zero8();
#pragma unroll 4
            for (int kk = 0; kk < DH; kk += 32) {
                FragBF a, b;
                a.u[0] = *(const uint4*)(qrow + kk + half * 8);       // LDS
                a.u[1] = *(const uint4*)(qrow + kk + 16 + half * 8);  // LDS
                b.u[0] = *(const uint4*)(krow + kk + half * 8);       // L2
                b.u[1] = *(const uint4*)(krow + kk + 16 + half * 8);
                s = wmma_bf16(a.v, b.v, s);
            }
#pragma unroll
            for (int g = 0; g < 8; ++g)
                S[mw * 16 + half * 8 + g][(jw * 2 + st) * 16 + ln] =
                    s[g] * scale;
        }
        __syncthreads();

        // ---- online softmax stats for row (mw*16 + ln), float4 scans
        const float*  srow = &S[mw * 16 + ln][0];
        const float4* s4   = (const float4*)srow;
        float cmax = -3.0e38f;
#pragma unroll 8
        for (int j = 0; j < 32; ++j) {
            const float4 x = s4[j];
            cmax = fmaxf(cmax, fmaxf(fmaxf(x.x, x.y), fmaxf(x.z, x.w)));
        }
        const float m_new = fmaxf(m_run, cmax);
        const float corr  = __expf(m_run - m_new);
        float ssum = 0.f;
#pragma unroll 8
        for (int j = 0; j < 32; ++j) {
            const float4 x = s4[j];
            ssum += __expf(x.x - m_new) + __expf(x.y - m_new) +
                    __expf(x.z - m_new) + __expf(x.w - m_new);
        }
        l_run = l_run * corr + ssum;
        m_run = m_new;

        // ---- rescale accumulators (row of acc element g = mw*16 + h*8 + g)
#pragma unroll
        for (int g = 0; g < 8; ++g) {
            const float cg = __shfl(corr, half * 8 + g, 32);
#pragma unroll
            for (int t = 0; t < 16; ++t) acc[t][g] *= cg;
        }

        // ---- build 4 bf16 P fragments (A rows = mw*16 + ln)
        FragBF P[4];
#pragma unroll
        for (int c4 = 0; c4 < 4; ++c4) {
#pragma unroll
            for (int i = 0; i < 8; ++i) {
                P[c4].e[i] =
                    (bf16)__expf(srow[c4 * 32 + half * 8 + i] - m_run);
                P[c4].e[i + 8] =
                    (bf16)__expf(srow[c4 * 32 + 16 + half * 8 + i] - m_run);
            }
        }

        // ---- acc += P @ V  (contiguous v^T rows, L2-resident)
#pragma unroll
        for (int t = 0; t < 16; ++t) {
            const bf16* vrow =
                vt + (long)(jw * 256 + t * 16 + ln) * NTOK + nc;
            if (nc + 128 < NTOK)
                __builtin_prefetch(vrow + 128, 0, 0);
#pragma unroll
            for (int c4 = 0; c4 < 4; ++c4) {
                FragBF bv;
                bv.u[0] = *(const uint4*)(vrow + c4 * 32 + half * 8);
                bv.u[1] = *(const uint4*)(vrow + c4 * 32 + 16 + half * 8);
                acc[t] = wmma_bf16(P[c4].v, bv.v, acc[t]);
            }
        }
    }

    // ---- epilogue: normalize by row sum, store fp32
#pragma unroll
    for (int g = 0; g < 8; ++g) {
        const float lg  = __shfl(l_run, half * 8 + g, 32);
        const float inv = 1.0f / lg;
        const long  row = block_m + mw * 16 + half * 8 + g;
#pragma unroll
        for (int t = 0; t < 16; ++t)
            out[row * DH + jw * 256 + t * 16 + ln] = acc[t][g] * inv;
    }
}

// ---------------------------------------------------------------------------
extern "C" void kernel_launch(void* const* d_in, const int* in_sizes, int n_in,
                              void* d_out, int out_size, void* d_ws,
                              size_t ws_size, hipStream_t stream)
{
    (void)in_sizes; (void)n_in; (void)out_size; (void)ws_size;
    const float* h  = (const float*)d_in[0];
    const float* wq = (const float*)d_in[1];
    const float* wk = (const float*)d_in[2];
    const float* wv = (const float*)d_in[3];

    // workspace: q (bf16 row-major), k (bf16 row-major), v^T (bf16) = 48 MB
    bf16* qb = (bf16*)d_ws;
    bf16* kb = qb + (size_t)NTOK * DH;
    bf16* vt = kb + (size_t)NTOK * DH;

    proj_kernel<<<dim3(NTOK / 16, 8, 3), dim3(256), 0, stream>>>(
        h, wq, wk, wv, qb, kb, vt);
    attn_kernel<<<dim3(NTOK / 32), dim3(256), 0, stream>>>(
        qb, kb, vt, (float*)d_out);
}
